// _UserModel_86165633892569
// MI455X (gfx1250) — matrix operand order repaired and checked by reading, not compiled, not run on hardware
//
#include <hip/hip_runtime.h>
#include <hip/hip_bf16.h>
#include <math.h>

// ---------------------------------------------------------------------------
// Problem constants (from reference)
// ---------------------------------------------------------------------------
#define BB      4096
#define LL      200
#define DD      64
#define NTOK    (BB * LL)          // 819200 tokens
#define NTILE   (NTOK / 16)        // 51200 16-token tiles
#define LSTM_L  4
#define EPSV    1e-10f

// Padded LDS row strides (bank-conflict avoidance on 64x4B banks):
//  - f16 tiles: 72 halves = 144B = 36 words -> 16 row starts hit 16 distinct banks
//  - f32 gates: 260 floats -> lanes 16-31 land on banks offset by 32 vs lanes 0-15
#define STH64   72     // f16 row stride for [16][64] tiles
#define STH128  136    // f16 row stride for [16][128] tiles
#define STG     260    // f32 row stride for [16][256] gates

typedef __attribute__((ext_vector_type(16))) _Float16 v16h;
typedef __attribute__((ext_vector_type(8)))  float    v8f;

// ---------------------------------------------------------------------------
// WMMA fragment loaders (contiguous k = 16*(lane/16) + i, same for A and B, wave32)
// ---------------------------------------------------------------------------
// A fragment (16x32 f16, M x K), source row-major [16][rowStride] in LDS.
//   lanes 0-15 : M=lane, halves 0..15 = K(kb+0..15)
//   lanes16-31 : M=lane-16, kb += 16
__device__ __forceinline__ v16h load_a_lds(const _Float16* X, int rowStride, int k_base, int lane) {
  int m  = lane & 15;
  int kb = k_base + ((lane & 16) ? 16 : 0);
  const _Float16* p = X + m * rowStride + kb;
  v16h a;
#pragma unroll
  for (int i = 0; i < 16; ++i) a[i] = p[i];
  return a;
}

// B fragment (32x16 f16, K x N) taken from row-major W[N][K] (i.e. B = W^T).
//   lanes 0-15 : N=lane, K = k_base+0..15  ;  lanes 16-31 : N=lane-16, K = k_base+16..31
//   -> each lane reads 16 contiguous halves of one W row (2x b128 loads).
__device__ __forceinline__ v16h load_b_wt(const _Float16* W, int ldk, int n_base, int k_base, int lane) {
  int n  = n_base + (lane & 15);
  int k0 = k_base + ((lane & 16) ? 16 : 0);
  const _Float16* p = W + n * ldk + k0;
  v16h b;
#pragma unroll
  for (int i = 0; i < 16; ++i) b[i] = p[i];
  return b;
}

__device__ __forceinline__ v8f wmma_f16(v16h a, v16h b, v8f c) {
  return __builtin_amdgcn_wmma_f32_16x16x32_f16(false, a, false, b, (short)0, c, false, false);
}

// OUT[16][64] (f16, LDS dst, leading dim dstLd; column offset folded into dst)
//   = act( A[16][K] @ W^T + bias ),  K in {64,128}, act: 0=none 1=leaky(0.2)
template<int K, int ACT>
__device__ __forceinline__ void gemm_tile64(const _Float16* A, int strideA,
                                            const _Float16* W, const float* bias,
                                            _Float16* dst, int dstLd, int lane) {
#pragma unroll
  for (int nt = 0; nt < 4; ++nt) {
    v8f acc = {};
#pragma unroll
    for (int kt = 0; kt < K / 32; ++kt) {
      v16h a = load_a_lds(A, strideA, kt * 32, lane);
      v16h b = load_b_wt(W, K, nt * 16, kt * 32, lane);
      acc = wmma_f16(a, b, acc);
    }
    int n  = nt * 16 + (lane & 15);
    int mb = (lane & 16) ? 8 : 0;
    float bv = bias[n];
#pragma unroll
    for (int r = 0; r < 8; ++r) {
      float v = acc[r] + bv;
      if (ACT == 1) v = (v > 0.f) ? v : 0.2f * v;
      dst[(mb + r) * dstLd + n] = (_Float16)v;
    }
  }
}

__device__ __forceinline__ float sigf(float x) { return 1.f / (1.f + __expf(-x)); }

// ---------------------------------------------------------------------------
// K0: convert weights f32 -> f16, combine LSTM biases
// ---------------------------------------------------------------------------
__global__ void k0_convert(const float* gv_W1, const float* gv_W2, const float* w1_W,
                           const float* att_W1, const float* Wih, const float* Whh,
                           const float* bih, const float* bhh,
                           _Float16* o_gv1, _Float16* o_gv2, _Float16* o_w1,
                           _Float16* o_att1, _Float16* o_ih, _Float16* o_hh,
                           float* o_bias) {
  int i = blockIdx.x * blockDim.x + threadIdx.x;
  int s = gridDim.x * blockDim.x;
  for (int j = i; j < 8192;  j += s) o_gv1[j]  = (_Float16)gv_W1[j];
  for (int j = i; j < 4096;  j += s) o_gv2[j]  = (_Float16)gv_W2[j];
  for (int j = i; j < 4096;  j += s) o_w1[j]   = (_Float16)w1_W[j];
  for (int j = i; j < 8192;  j += s) o_att1[j] = (_Float16)att_W1[j];
  for (int j = i; j < 65536; j += s) o_ih[j]   = (_Float16)Wih[j];
  for (int j = i; j < 65536; j += s) o_hh[j]   = (_Float16)Whh[j];
  for (int j = i; j < 1024;  j += s) o_bias[j] = bih[j] + bhh[j];
}

// ---------------------------------------------------------------------------
// K1: per-token MLP + attention logits. One wave == one 16-token tile.
//     block = 128 threads (4 waves), grid = NTILE/4
// ---------------------------------------------------------------------------
#define K1_WAVES 4
__global__ void k1_token(const int* __restrict__ u_item_pad, const int* __restrict__ uids,
                         const float* __restrict__ item_table, const float* __restrict__ rate_table,
                         const float* __restrict__ user_table,
                         const _Float16* __restrict__ hW_gv1, const float* __restrict__ gv_b1,
                         const _Float16* __restrict__ hW_gv2, const float* __restrict__ gv_b2,
                         const _Float16* __restrict__ hW_w1,  const float* __restrict__ w1_b,
                         const _Float16* __restrict__ hW_att1,const float* __restrict__ att_b1,
                         const float* __restrict__ att_W2, const float* __restrict__ att_b2,
                         _Float16* __restrict__ xij, float* __restrict__ logits) {
  __shared__ _Float16 sA[K1_WAVES][16 * STH128]; // concat(q,e) then concat(w1x,w1p)
  __shared__ _Float16 sB[K1_WAVES][16 * STH64];  // h1 then hA
  __shared__ _Float16 sX[K1_WAVES][16 * STH64];  // x_ij
  __shared__ _Float16 sP[K1_WAVES][16 * STH64];  // p_i (masked user emb)
  __shared__ float    sMask[K1_WAVES][16];

  int tid  = threadIdx.x;
  int lane = tid & 31;
  int wv   = tid >> 5;
  _Float16* bA = sA[wv];
  _Float16* bB = sB[wv];
  _Float16* bX = sX[wv];
  _Float16* bP = sP[wv];

  int tile = blockIdx.x * K1_WAVES + wv;
  int tok0 = tile * 16;

  // ---- gather: lanes pair up (2 lanes / token) ----
  {
    int m    = lane >> 1;
    int half = lane & 1;
    int tok  = tok0 + m;
    int b    = tok / LL;
    int item = u_item_pad[tok * 2 + 0];
    int rate = u_item_pad[tok * 2 + 1];
    float mk = (item > 0) ? 1.f : 0.f;
    int uid  = uids[b];
    if (half == 0) {
      sMask[wv][m] = mk;
#pragma unroll 8
      for (int j = 0; j < 64; ++j) bA[m * STH128 + j] = (_Float16)item_table[item * 64 + j];
    } else {
#pragma unroll 8
      for (int j = 0; j < 64; ++j) bA[m * STH128 + 64 + j] = (_Float16)rate_table[rate * 64 + j];
    }
#pragma unroll 8
    for (int j = 0; j < 32; ++j) {
      int d = half * 32 + j;
      bP[m * STH64 + d] = (_Float16)(user_table[(size_t)uid * 64 + d] * mk);
    }
  }
  __syncthreads();

  // h1 = leaky( [q,e] @ gv_W1^T + b1 )
  gemm_tile64<128, 1>(bA, STH128, hW_gv1, gv_b1, bB, STH64, lane);
  __syncthreads();
  // x_ij = h1 @ gv_W2^T + b2
  gemm_tile64<64, 0>(bB, STH64, hW_gv2, gv_b2, bX, STH64, lane);
  __syncthreads();

  // spill x_ij tile to global (f16, row-major [tok][64]) : each lane one half-row
  {
    int m  = lane >> 1;
    int c0 = (lane & 1) * 32;
    _Float16* dst = xij + ((size_t)(tok0 + m)) * 64 + c0;
#pragma unroll 8
    for (int j = 0; j < 32; ++j) dst[j] = bX[m * STH64 + c0 + j];
  }

  // w1x -> att_in[:, 0:64] ; w1p -> att_in[:, 64:128]
  gemm_tile64<64, 0>(bX, STH64, hW_w1, w1_b, bA, STH128, lane);
  gemm_tile64<64, 0>(bP, STH64, hW_w1, w1_b, bA + 64, STH128, lane);
  __syncthreads();
  // hA = leaky( att_in @ att_W1^T + att_b1 )
  gemm_tile64<128, 1>(bA, STH128, hW_att1, att_b1, bB, STH64, lane);
  __syncthreads();

  // logit -> exp(logit)*mask : split dot across lane halves, shfl-xor reduce (wave32)
  {
    int m  = lane & 15;
    int j0 = (lane & 16) ? 32 : 0;
    float s = 0.f;
#pragma unroll 8
    for (int j = 0; j < 32; ++j) s += (float)bB[m * STH64 + j0 + j] * att_W2[j0 + j];
    s += __shfl_xor(s, 16);
    if (lane < 16) logits[tok0 + m] = __expf(s + att_b2[0]) * sMask[wv][m];
  }
}

// ---------------------------------------------------------------------------
// K2: attention pooling + agg matvec. One block per batch row.
// ---------------------------------------------------------------------------
__global__ void k2_pool(const float* __restrict__ logits, const _Float16* __restrict__ xij,
                        const float* __restrict__ agg_W, const float* __restrict__ agg_b,
                        float* __restrict__ hiL) {
  __shared__ float sw[256];
  __shared__ float pp[4][64];
  __shared__ float psh[64];
  int b = blockIdx.x, tid = threadIdx.x;

  float w = (tid < LL) ? logits[(size_t)b * LL + tid] : 0.f;
  sw[tid] = w;
  __syncthreads();
  for (int s = 128; s > 0; s >>= 1) {
    if (tid < s) sw[tid] += sw[tid + s];
    __syncthreads();
  }

  int d  = tid & 63;
  int cs = (tid >> 6) * 50;
  float acc = 0.f;
  for (int i = 0; i < 50; ++i) {
    int l = cs + i;
    float wl = logits[(size_t)b * LL + l];
    acc += wl * (float)xij[((size_t)b * LL + l) * 64 + d];
  }
  pp[tid >> 6][d] = acc;
  __syncthreads();
  if (tid < 64)
    psh[tid] = (pp[0][tid] + pp[1][tid] + pp[2][tid] + pp[3][tid]) / (sw[0] + EPSV);
  __syncthreads();
  if (tid < 64) {
    float a = agg_b[tid];
#pragma unroll 8
    for (int k = 0; k < 64; ++k) a += psh[k] * agg_W[tid * 64 + k];
    hiL[(size_t)b * 64 + tid] = a > 0.f ? a : 0.f;
  }
}

// ---------------------------------------------------------------------------
// K3: 4-layer LSTM. One block owns 16 batch rows for ALL layers/timesteps.
//     block = 256 (8 waves, each wave -> 32 gate columns), grid = BB/16.
//     Gate GEMM [16,256] = Xt@Wih^T + H@Whh^T via v_wmma_f32_16x16x32_f16,
//     B fragments register-resident per layer, Xt register-prefetched one
//     step ahead (2 barriers / step, global latency fully hidden).
// ---------------------------------------------------------------------------
__global__ void k3_lstm(const _Float16* __restrict__ hW_ih, const _Float16* __restrict__ hW_hh,
                        const float* __restrict__ biasComb,
                        _Float16* __restrict__ xijBuf, _Float16* __restrict__ seqA,
                        const float* __restrict__ hiL, float* __restrict__ out) {
  __shared__ _Float16 sXt[16 * STH64];
  __shared__ _Float16 sH[16 * STH64];
  __shared__ float    sC[16 * 64];
  __shared__ float    sG[16 * STG];
  __shared__ float    sBias[256];

  int tid  = threadIdx.x;
  int lane = tid & 31;
  int wv   = tid >> 5;
  int b0   = blockIdx.x * 16;

  // per-thread Xt staging slot: 4 halves (8B, aligned: STH64*2=144B rows, c0*2 8B-mult)
  int pm = tid >> 4;          // 0..15  (batch row within tile)
  int pc = (tid & 15) * 4;    // 0..60  (column)

  for (int l = 0; l < LSTM_L; ++l) {
    const _Float16* Wih = hW_ih + l * 16384;
    const _Float16* Whh = hW_hh + l * 16384;
    const _Float16* inS = (l & 1) ? seqA : xijBuf; // 0:xij->seqA 1:seqA->xij 2:xij->seqA 3:seqA->out
    _Float16* outS      = (l == 3) ? nullptr : ((l & 1) ? xijBuf : seqA);

    sBias[tid] = biasComb[l * 256 + tid];
    for (int i = tid; i < 16 * 64; i += 256) sC[i] = 0.f;
    for (int i = tid; i < 16 * STH64; i += 256) sH[i] = (_Float16)0.f;

    // register-resident B fragments for this wave's 32 gate columns
    v16h Bih[2][2], Bhh[2][2];
#pragma unroll
    for (int nt = 0; nt < 2; ++nt) {
      int n0 = wv * 32 + nt * 16;
#pragma unroll
      for (int kt = 0; kt < 2; ++kt) {
        Bih[nt][kt] = load_b_wt(Wih, 64, n0, kt * 32, lane);
        Bhh[nt][kt] = load_b_wt(Whh, 64, n0, kt * 32, lane);
      }
    }

    const _Float16* myRow = inS + ((size_t)(b0 + pm) * LL) * 64 + pc;
    // stage t=0
    {
      uint2 pre = *(const uint2*)(myRow);
      *(uint2*)&sXt[pm * STH64 + pc] = pre;
    }
    __syncthreads();

    for (int t = 0; t < LL; ++t) {
      // prefetch Xt(t+1) into registers (latency overlapped with WMMA phase)
      uint2 pre;
      if (t + 1 < LL) pre = *(const uint2*)(myRow + (size_t)(t + 1) * 64);

      // gates = Xt@Wih^T + H@Whh^T  (per wave: 2 n-tiles x 4 WMMAs)
#pragma unroll
      for (int nt = 0; nt < 2; ++nt) {
        v8f acc = {};
#pragma unroll
        for (int kt = 0; kt < 2; ++kt) {
          v16h ax = load_a_lds(sXt, STH64, kt * 32, lane);
          acc = wmma_f16(ax, Bih[nt][kt], acc);
          v16h ah = load_a_lds(sH, STH64, kt * 32, lane);
          acc = wmma_f16(ah, Bhh[nt][kt], acc);
        }
        int n  = wv * 32 + nt * 16 + (lane & 15);
        int mb = (lane & 16) ? 8 : 0;
#pragma unroll
        for (int r = 0; r < 8; ++r) sG[(mb + r) * STG + n] = acc[r];
      }
      __syncthreads();   // gates visible; all sXt/sH reads complete

      // fused cell update (4 elements / thread)
#pragma unroll
      for (int q = 0; q < 4; ++q) {
        int e = tid + q * 256;
        int m = e >> 6, j = e & 63;
        float gi = sG[m * STG + j]       + sBias[j];
        float gf = sG[m * STG + 64 + j]  + sBias[64 + j];
        float gg = sG[m * STG + 128 + j] + sBias[128 + j];
        float go = sG[m * STG + 192 + j] + sBias[192 + j];
        float c  = sigf(gf) * sC[m * 64 + j] + sigf(gi) * tanhf(gg);
        float h  = sigf(go) * tanhf(c);
        sC[m * 64 + j] = c;
        sH[m * STH64 + j] = (_Float16)h;
        if (outS) {
          outS[((size_t)(b0 + m) * LL + t) * 64 + j] = (_Float16)h;
        } else if (t == LL - 1) {
          out[(size_t)(b0 + m) * 64 + j] = h * hiL[(size_t)(b0 + m) * 64 + j];
        }
      }
      // stage prefetched Xt(t+1) (sXt reads finished at the barrier above)
      if (t + 1 < LL) *(uint2*)&sXt[pm * STH64 + pc] = pre;
      __syncthreads();   // sH / sXt ready for next step
    }
    __syncthreads();
  }
}

// ---------------------------------------------------------------------------
// Host launcher
// ---------------------------------------------------------------------------
extern "C" void kernel_launch(void* const* d_in, const int* in_sizes, int n_in,
                              void* d_out, int out_size, void* d_ws, size_t ws_size,
                              hipStream_t stream) {
  (void)in_sizes; (void)n_in; (void)out_size; (void)ws_size;
  const int*   uids       = (const int*)d_in[0];
  const int*   u_item_pad = (const int*)d_in[1];
  /* d_in[2] soc_edge_index unused by reference */
  const float* user_table = (const float*)d_in[3];
  const float* item_table = (const float*)d_in[4];
  const float* rate_table = (const float*)d_in[5];
  const float* w1_W   = (const float*)d_in[6];
  const float* w1_b   = (const float*)d_in[7];
  const float* gv_W1  = (const float*)d_in[8];
  const float* gv_b1  = (const float*)d_in[9];
  const float* gv_W2  = (const float*)d_in[10];
  const float* gv_b2  = (const float*)d_in[11];
  const float* att_W1 = (const float*)d_in[12];
  const float* att_b1 = (const float*)d_in[13];
  const float* att_W2 = (const float*)d_in[14];
  const float* att_b2 = (const float*)d_in[15];
  const float* agg_W  = (const float*)d_in[16];
  const float* agg_b  = (const float*)d_in[17];
  const float* lWih   = (const float*)d_in[18];
  const float* lWhh   = (const float*)d_in[19];
  const float* lbih   = (const float*)d_in[20];
  const float* lbhh   = (const float*)d_in[21];
  float* out = (float*)d_out;

  // ---- workspace carve (256B aligned) ----
  char* ws = (char*)d_ws;
  size_t off = 0;
  auto carve = [&](size_t bytes) -> char* {
    char* p = ws + off;
    off = (off + bytes + 255) & ~(size_t)255;
    return p;
  };
  _Float16* hW_gv1  = (_Float16*)carve(8192 * 2);
  _Float16* hW_gv2  = (_Float16*)carve(4096 * 2);
  _Float16* hW_w1   = (_Float16*)carve(4096 * 2);
  _Float16* hW_att1 = (_Float16*)carve(8192 * 2);
  _Float16* hW_ih   = (_Float16*)carve(65536 * 2);
  _Float16* hW_hh   = (_Float16*)carve(65536 * 2);
  float*    biasC   = (float*)carve(1024 * 4);
  float*    hiL     = (float*)carve((size_t)BB * 64 * 4);
  float*    logits  = (float*)carve((size_t)NTOK * 4);
  _Float16* xij     = (_Float16*)carve((size_t)NTOK * 64 * 2);
  _Float16* seqA    = (_Float16*)carve((size_t)NTOK * 64 * 2);

  k0_convert<<<256, 256, 0, stream>>>(gv_W1, gv_W2, w1_W, att_W1, lWih, lWhh, lbih, lbhh,
                                      hW_gv1, hW_gv2, hW_w1, hW_att1, hW_ih, hW_hh, biasC);

  k1_token<<<NTILE / K1_WAVES, 32 * K1_WAVES, 0, stream>>>(
      u_item_pad, uids, item_table, rate_table, user_table,
      hW_gv1, gv_b1, hW_gv2, gv_b2, hW_w1, w1_b, hW_att1, att_b1,
      att_W2, att_b2, xij, logits);

  k2_pool<<<BB, 256, 0, stream>>>(logits, xij, agg_W, agg_b, hiL);

  k3_lstm<<<BB / 16, 256, 0, stream>>>(hW_ih, hW_hh, biasC, xij, seqA, hiL, out);
}
